// Transformer_20177756356752
// MI455X (gfx1250) — matrix operand order repaired and checked by reading, not compiled, not run on hardware
//
#include <hip/hip_runtime.h>
#include <math.h>
#include <stdint.h>

// ---------------- problem constants ----------------
static constexpr int TB = 2, TS = 2048, TD = 768, TH = 12, TV = 50257, TF = 3072;
static constexpr int THD = 64;                 // head dim
static constexpr int TM  = TB * TS;            // 4096 rows of activations
static constexpr int TVPAD = 50432;            // 197 * 256

// ---------------- CDNA5 WMMA types ----------------
typedef __attribute__((ext_vector_type(16))) __bf16 v16bf;
typedef __attribute__((ext_vector_type(8)))  __bf16 v8bf;
typedef __attribute__((ext_vector_type(8)))  float  v8f;

union V16U { v16bf v; v8bf h[2]; };

// ---------------- TDM availability ----------------
#if defined(__has_builtin)
#if __has_builtin(__builtin_amdgcn_tensor_load_to_lds) && \
    __has_builtin(__builtin_amdgcn_s_wait_tensorcnt)
#define USE_TDM 1
#endif
#endif

#ifdef USE_TDM
typedef __attribute__((ext_vector_type(4))) unsigned int u32x4;
typedef __attribute__((ext_vector_type(4))) int          i32x4;
typedef __attribute__((ext_vector_type(8))) int          i32x8;

static constexpr int AROW = 40;   // LDS row stride in bf16 elems (64B data + 16B pad)

// Issue a TDM load of a 32x32 bf16 tile (rows x K-elems) into LDS.
// D# bit layout per CDNA5 ISA ch.8 (group0 128b, group1 256b).
__device__ __forceinline__ void tdm_load_tile_a(unsigned lds_off,
                                                const __bf16* gaddr, int K) {
    unsigned long long ga = (unsigned long long)(uintptr_t)gaddr;
    u32x4 g0;
    g0[0] = 1u;                                             // count=1 (valid user D#)
    g0[1] = lds_off;                                        // lds_addr (bytes)
    g0[2] = (unsigned)(ga & 0xFFFFFFFFu);                   // global_addr[31:0]
    g0[3] = (unsigned)((ga >> 32) & 0x01FFFFFFu) | (2u << 30); // ga[56:32] | type=2
    unsigned td0 = (unsigned)K;       // tensor dim0 (elements)
    unsigned td1 = 32u;               // tensor dim1 (rows covered)
    i32x8 g1;
    g1[0] = (int)((1u << 16)          // data_size = 1 -> 2 bytes
                | (1u << 20)          // pad_enable
                | (3u << 22)          // pad_interval code 3 -> every 64B
                | (3u << 25));        // pad_amount  code 3 -> 16B
    g1[1] = (int)((td0 & 0xFFFFu) << 16);                   // tensor_dim0[15:0]
    g1[2] = (int)(((td0 >> 16) & 0xFFFFu) | ((td1 & 0xFFFFu) << 16));
    g1[3] = (int)(((td1 >> 16) & 0xFFFFu) | (32u << 16));   // tile_dim0 = 32
    g1[4] = (int)(32u);                                     // tile_dim1 = 32, tile_dim2=0
    g1[5] = (int)td0;                                       // tensor_dim0_stride[31:0] = K
    g1[6] = 0;
    g1[7] = 0;
    i32x4 gz = {0, 0, 0, 0};
#if __clang_major__ >= 23
    i32x8 gz8 = {0, 0, 0, 0, 0, 0, 0, 0};
    __builtin_amdgcn_tensor_load_to_lds(g0, g1, gz, gz, gz8, 0);
#else
    __builtin_amdgcn_tensor_load_to_lds(g0, g1, gz, gz, 0);
#endif
}
#endif // USE_TDM

// load 16 contiguous bf16 (16B-aligned halves -> 2x b128 loads; works global or LDS)
__device__ __forceinline__ v16bf load_bf16x16(const __bf16* p) {
    V16U t;
    t.h[0] = *(const v8bf*)(p);
    t.h[1] = *(const v8bf*)(p + 8);
    return t.v;
}

// A-fragment (16x32, 16-bit): elements 0..15 <- K = 2*kb8..2*kb8+15
// (contiguous; kb8 = 0 for lanes 0-15, 8 for 16-31)
__device__ __forceinline__ v16bf load_afrag(const __bf16* rowp, int kb8) {
    V16U t;
    t.h[0] = *(const v8bf*)(rowp + 2 * kb8);
    t.h[1] = *(const v8bf*)(rowp + 2 * kb8 + 8);
    return t.v;
}

__device__ __forceinline__ v8f zero8() {
    v8f v = {0.f, 0.f, 0.f, 0.f, 0.f, 0.f, 0.f, 0.f};
    return v;
}

__device__ __forceinline__ v8f wmma_bf16(v16bf a, v16bf b, v8f c) {
    return __builtin_amdgcn_wmma_f32_16x16x32_bf16(false, a, false, b,
                                                   (short)0, c, false, false);
}

// ---------------- weight prep kernels ----------------
__global__ __launch_bounds__(256) void wtrans_kernel(const float* __restrict__ W,
                                                     __bf16* __restrict__ BT,
                                                     int K, int N, int Npad) {
    size_t idx = (size_t)blockIdx.x * 256 + threadIdx.x;
    if (idx >= (size_t)Npad * K) return;
    int n = (int)(idx / K);
    int k = (int)(idx % K);
    float v = (n < N) ? W[(size_t)k * N + n] : 0.f;
    BT[idx] = (__bf16)v;
}

// WoT[n, k] = sum_j W_o[(j*D + k), n]   (tile(out,H) @ W_o == out @ sum_j block_j)
__global__ __launch_bounds__(256) void wosum_kernel(const float* __restrict__ Wo,
                                                    __bf16* __restrict__ WoT) {
    int idx = blockIdx.x * 256 + threadIdx.x;   // over D*D
    int n = idx / TD, k = idx % TD;
    float s = 0.f;
    for (int j = 0; j < TH; ++j)
        s += Wo[((size_t)(j * TD + k)) * TD + n];
    WoT[(size_t)n * TD + k] = (__bf16)s;
}

// ---------------- embedding + positional encoding ----------------
__global__ __launch_bounds__(256) void embed_kernel(const int* __restrict__ inputs,
                                                    const float* __restrict__ emb,
                                                    float* __restrict__ xf,
                                                    __bf16* __restrict__ xbf) {
    size_t idx = (size_t)blockIdx.x * 256 + threadIdx.x;   // over B*S*D
    int d = (int)(idx % TD);
    int s = (int)((idx / TD) % TS);
    int b = (int)(idx / ((size_t)TD * TS));
    int tok = inputs[b * TS + s];
    float freq = __powf(10000.0f, -(float)(2 * d) / (float)TD);
    float div  = (float)s * freq;
    float pe   = ((d & 1) == 0) ? __sinf(div) : __cosf(div);
    float v = emb[(size_t)tok * TD + d] + pe;
    xf[idx]  = v;
    xbf[idx] = (__bf16)v;
}

// ---------------- split qkv -> q (scaled), k, vT ----------------
__global__ __launch_bounds__(256) void split_qkv_kernel(const float* __restrict__ qkv,
                                                        __bf16* __restrict__ q,
                                                        __bf16* __restrict__ k,
                                                        __bf16* __restrict__ vT) {
    size_t idx = (size_t)blockIdx.x * 256 + threadIdx.x;   // over B*S*D
    int d = (int)(idx % TD);
    int s = (int)((idx / TD) % TS);
    int b = (int)(idx / ((size_t)TD * TS));
    int h = d >> 6, hd = d & 63;
    const float* base = qkv + (size_t)(b * TS + s) * (3 * TD);
    size_t qi = ((size_t)(b * TH + h) * TS + s) * THD + hd;
    q[qi] = (__bf16)(base[d] * 0.125f);            // 1/sqrt(64) folded in
    k[qi] = (__bf16)(base[TD + d]);
    vT[((size_t)(b * TH + h) * THD + hd) * TS + s] = (__bf16)(base[2 * TD + d]);
}

// ---------------- generic bf16 WMMA GEMM: C = A[M,K] @ BT[N,K]^T ----------------
// wave: 32x64 tile (2 M-frags x 4 N-frags); block: 8 waves -> 64x256.
// A staged to LDS by the Tensor Data Mover; K-loop manually unrolled x2 into
// explicit ping-pong phases so the TDM buffers and the B-fragment register
// sets alternate roles without loop-carried register copies.
// NOTE: all call sites use K % 64 == 0 (K = 768 or 3072).
__global__ __launch_bounds__(256) void gemm_bf16_kernel(
    const __bf16* __restrict__ A, const __bf16* __restrict__ BT,
    const float* __restrict__ bias, const float* __restrict__ resid,
    float* __restrict__ Cf, __bf16* __restrict__ Cbf,
    int M, int N, int K, int relu)
{
#ifdef USE_TDM
    __shared__ __attribute__((aligned(64))) __bf16 atile[8][2][32 * AROW];
#endif
    int lane = threadIdx.x & 31, wid = threadIdx.x >> 5;
    int m    = lane & 15, half = lane >> 4;
    int mi   = wid >> 2, ni = wid & 3;
    int m0   = blockIdx.y * 64 + mi * 32;
    int n0   = blockIdx.x * 256 + ni * 64;
    int kb8  = half * 8, kb16 = half * 16;

    v8f c[2][4];
#pragma unroll
    for (int mm = 0; mm < 2; ++mm)
#pragma unroll
        for (int nn = 0; nn < 4; ++nn) c[mm][nn] = zero8();

    const __bf16* Abase = A + (size_t)m0 * K;   // wave's 32-row panel

    auto loadB = [&](int kk, v16bf* bf) {
#pragma unroll
        for (int nn = 0; nn < 4; ++nn) {
            const __bf16* br = BT + (size_t)(n0 + nn * 16 + m) * K + kk + kb16;
            bf[nn] = load_bf16x16(br);
            __builtin_prefetch(br + 64, 0, 3);   // two K-steps ahead, near-cache
        }
    };

#ifdef USE_TDM
    unsigned lds0 = (unsigned)(uintptr_t)&atile[wid][0][0];
    unsigned lds1 = (unsigned)(uintptr_t)&atile[wid][1][0];
    tdm_load_tile_a(lds0, Abase, K);            // prime buffer 0 (kk = 0)
#endif
    v16bf b0[4], b1[4];
    loadB(0, b0);                               // prime B registers (kk = 0)

    for (int kk = 0; kk < K; kk += 64) {
        v16bf a[2];

        // ---- phase 0: consume (buf0, b0); prefetch (buf1, b1) for kk+32 ----
#ifdef USE_TDM
        // buf1 was read in the previous iteration's phase 1
        asm volatile("s_wait_dscnt 0x0" ::: "memory");
        tdm_load_tile_a(lds1, Abase + kk + 32, K);
#endif
        loadB(kk + 32, b1);                     // kk+32 < K always (K % 64 == 0)
#ifdef USE_TDM
        __builtin_amdgcn_s_wait_tensorcnt(1);   // buf0 ready, buf1 in flight
        {
            const __bf16* l0 = &atile[wid][0][0];
#pragma unroll
            for (int mm = 0; mm < 2; ++mm)
                a[mm] = load_afrag(l0 + (mm * 16 + m) * AROW, kb8);
        }
#else
#pragma unroll
        for (int mm = 0; mm < 2; ++mm)
            a[mm] = load_afrag(A + (size_t)(m0 + mm * 16 + m) * K + kk, kb8);
#endif
#pragma unroll
        for (int mm = 0; mm < 2; ++mm)
#pragma unroll
            for (int nn = 0; nn < 4; ++nn)
                c[mm][nn] = wmma_bf16(a[mm], b0[nn], c[mm][nn]);

        // ---- phase 1: consume (buf1, b1); prefetch (buf0, b0) for kk+64 ----
        bool more = (kk + 64 < K);
        if (more) {
#ifdef USE_TDM
            asm volatile("s_wait_dscnt 0x0" ::: "memory");
            tdm_load_tile_a(lds0, Abase + kk + 64, K);
#endif
            loadB(kk + 64, b0);
        }
#ifdef USE_TDM
        if (more) __builtin_amdgcn_s_wait_tensorcnt(1);   // buf1 ready
        else      __builtin_amdgcn_s_wait_tensorcnt(0);
        {
            const __bf16* l1 = &atile[wid][1][0];
#pragma unroll
            for (int mm = 0; mm < 2; ++mm)
                a[mm] = load_afrag(l1 + (mm * 16 + m) * AROW, kb8);
        }
#else
#pragma unroll
        for (int mm = 0; mm < 2; ++mm)
            a[mm] = load_afrag(A + (size_t)(m0 + mm * 16 + m) * K + kk + 32, kb8);
#endif
#pragma unroll
        for (int mm = 0; mm < 2; ++mm)
#pragma unroll
            for (int nn = 0; nn < 4; ++nn)
                c[mm][nn] = wmma_bf16(a[mm], b1[nn], c[mm][nn]);
    }

#pragma unroll
    for (int mm = 0; mm < 2; ++mm)
#pragma unroll
        for (int nn = 0; nn < 4; ++nn) {
            int col = n0 + nn * 16 + m;
            if (col < N) {
                float bv = bias ? bias[col] : 0.f;
#pragma unroll
                for (int r = 0; r < 8; ++r) {
                    int row = m0 + mm * 16 + half * 8 + r;   // C layout: half -> M +8
                    float v = c[mm][nn][r] + bv;
                    size_t o = (size_t)row * N + col;
                    if (resid) v += resid[o];
                    if (relu)  v = fmaxf(v, 0.f);
                    if (Cf)  Cf[o]  = v;
                    if (Cbf) Cbf[o] = (__bf16)v;
                }
            }
        }
}

// ---------------- flash attention (causal), one wave per 16 q-rows ----------------
__global__ __launch_bounds__(256) void attention_kernel(const __bf16* __restrict__ q,
                                                        const __bf16* __restrict__ k,
                                                        const __bf16* __restrict__ vT,
                                                        __bf16* __restrict__ out) {
    __shared__ __attribute__((aligned(64))) __bf16 pbuf[8][16 * 32];

    int lane = threadIdx.x & 31, wid = threadIdx.x >> 5;
    int m    = lane & 15, half = lane >> 4;
    int b = blockIdx.z, h = blockIdx.y;
    int q0 = (blockIdx.x * 8 + wid) * 16;

    const __bf16* qbase = q  + ((size_t)(b * TH + h) * TS + q0) * THD;
    const __bf16* kbase = k  + ((size_t)(b * TH + h)) * (size_t)TS * THD;
    const __bf16* vbase = vT + ((size_t)(b * TH + h)) * (size_t)THD * TS;

    int kb8 = half * 8, kb16 = half * 16;

    // Q fragments (head dim 64 -> two k=32 fragments), reused every tile
    const __bf16* qrow = qbase + (size_t)m * THD;
    v16bf qf0 = load_afrag(qrow, kb8);
    v16bf qf1 = load_afrag(qrow + 32, kb8);

    float mrow[8], lrow[8];
    v8f o[4];
#pragma unroll
    for (int r = 0; r < 8; ++r) { mrow[r] = -1e30f; lrow[r] = 0.f; }
#pragma unroll
    for (int nh = 0; nh < 4; ++nh) o[nh] = zero8();

    int kend = q0 + 16;                        // causal: cols < kend
    for (int kt = 0; kt < kend; kt += 32) {
        // ---- scores: S = Q @ K^T  (two 16-col tiles) ----
        const __bf16* kr0 = kbase + (size_t)(kt + m) * THD;
        const __bf16* kr1 = kbase + (size_t)(kt + 16 + m) * THD;
        v16bf b00 = load_bf16x16(kr0 + kb16);
        v16bf b01 = load_bf16x16(kr0 + 32 + kb16);
        v16bf b10 = load_bf16x16(kr1 + kb16);
        v16bf b11 = load_bf16x16(kr1 + 32 + kb16);

        v8f s0 = zero8(), s1 = zero8();
        s0 = wmma_bf16(qf0, b00, s0); s0 = wmma_bf16(qf1, b01, s0);
        s1 = wmma_bf16(qf0, b10, s1); s1 = wmma_bf16(qf1, b11, s1);

        // ---- causal mask + online softmax (row reductions within lane halves) ----
#pragma unroll
        for (int r = 0; r < 8; ++r) {
            int row = q0 + half * 8 + r;
            float x0 = (kt + m      <= row) ? s0[r] : -1e30f;
            float x1 = (kt + 16 + m <= row) ? s1[r] : -1e30f;

            float t = fmaxf(x0, x1);
#pragma unroll
            for (int msk = 1; msk < 16; msk <<= 1)
                t = fmaxf(t, __shfl_xor(t, msk, 16));
            float mn = fmaxf(mrow[r], t);
            float alpha = __expf(mrow[r] - mn);
            mrow[r] = mn;

            float e0 = __expf(x0 - mn);
            float e1 = __expf(x1 - mn);
            float rs = e0 + e1;
#pragma unroll
            for (int msk = 1; msk < 16; msk <<= 1)
                rs += __shfl_xor(rs, msk, 16);
            lrow[r] = lrow[r] * alpha + rs;

#pragma unroll
            for (int nh = 0; nh < 4; ++nh) o[nh][r] *= alpha;

            int lr = half * 8 + r;             // local row in P tile
            pbuf[wid][lr * 32 + m]      = (__bf16)e0;
            pbuf[wid][lr * 32 + 16 + m] = (__bf16)e1;
        }

        // wave-local LDS visibility before re-reading in A-fragment layout
        asm volatile("s_wait_dscnt 0x0" ::: "memory");

        // ---- P (16x32) as A-fragment, accumulate O += P @ V ----
        const __bf16* prow = &pbuf[wid][m * 32];
        V16U pa;
        pa.h[0] = *(const v8bf*)(prow + 2 * kb8);
        pa.h[1] = *(const v8bf*)(prow + 2 * kb8 + 8);

#pragma unroll
        for (int nh = 0; nh < 4; ++nh) {
            const __bf16* vr = vbase + (size_t)(nh * 16 + m) * TS + kt + kb16;
            v16bf vf = load_bf16x16(vr);
            o[nh] = wmma_bf16(pa.v, vf, o[nh]);
        }
    }

    // ---- normalize and write [B,S,D] bf16 (heads interleaved) ----
#pragma unroll
    for (int r = 0; r < 8; ++r) {
        int row = q0 + half * 8 + r;
        float inv = 1.0f / lrow[r];
#pragma unroll
        for (int nh = 0; nh < 4; ++nh) {
            int col = nh * 16 + m;
            out[((size_t)b * TS + row) * TD + h * THD + col] = (__bf16)(o[nh][r] * inv);
        }
    }
}

// ---------------- layernorm (one block per row) ----------------
__global__ __launch_bounds__(256) void layernorm_kernel(const float* __restrict__ x,
                                                        const float* __restrict__ g,
                                                        const float* __restrict__ bb,
                                                        float* __restrict__ yf,
                                                        __bf16* __restrict__ ybf) {
    __shared__ float rs[256], rs2[256];
    int row = blockIdx.x;
    const float* xr = x + (size_t)row * TD;
    float s = 0.f, s2 = 0.f;
    for (int i = threadIdx.x; i < TD; i += 256) { float v = xr[i]; s += v; s2 += v * v; }
    rs[threadIdx.x] = s; rs2[threadIdx.x] = s2;
    __syncthreads();
    for (int o = 128; o > 0; o >>= 1) {
        if (threadIdx.x < o) { rs[threadIdx.x] += rs[threadIdx.x + o]; rs2[threadIdx.x] += rs2[threadIdx.x + o]; }
        __syncthreads();
    }
    float mean = rs[0] / (float)TD;
    float var  = rs2[0] / (float)TD - mean * mean;
    float rstd = rsqrtf(var + 1e-5f);
    for (int i = threadIdx.x; i < TD; i += 256) {
        float v = (xr[i] - mean) * rstd * g[i] + bb[i];
        if (yf) yf[(size_t)row * TD + i] = v;
        ybf[(size_t)row * TD + i] = (__bf16)v;
    }
}

// ---------------- host orchestration ----------------
extern "C" void kernel_launch(void* const* d_in, const int* in_sizes, int n_in,
                              void* d_out, int out_size, void* d_ws, size_t ws_size,
                              hipStream_t stream) {
    (void)in_sizes; (void)n_in; (void)out_size; (void)ws_size;
    const int*   inputs = (const int*)  d_in[0];
    const float* emb    = (const float*)d_in[1];
    const float* W_qkv  = (const float*)d_in[2];
    const float* b_qkv  = (const float*)d_in[3];
    const float* W_o    = (const float*)d_in[4];
    const float* b_o    = (const float*)d_in[5];
    const float* ln1_g  = (const float*)d_in[6];
    const float* ln1_b  = (const float*)d_in[7];
    const float* W_ff1  = (const float*)d_in[8];
    const float* b_ff1  = (const float*)d_in[9];
    const float* W_ff2  = (const float*)d_in[10];
    const float* b_ff2  = (const float*)d_in[11];
    const float* ln2_g  = (const float*)d_in[12];
    const float* ln2_b  = (const float*)d_in[13];
    const float* W_out  = (const float*)d_in[14];
    const float* b_out  = (const float*)d_in[15];
    float* out = (float*)d_out;

    // workspace carve-out (256B aligned)
    char* ws = (char*)d_ws;
    size_t off = 0;
    auto alloc = [&](size_t bytes) -> char* {
        char* p = ws + off;
        off = (off + bytes + 255) & ~(size_t)255;
        return p;
    };
    float*  x_f32   = (float*) alloc((size_t)TM * TD * 4);
    __bf16* x_bf    = (__bf16*)alloc((size_t)TM * TD * 2);
    __bf16* WqkvT   = (__bf16*)alloc((size_t)(3 * TD) * TD * 2);
    __bf16* WoT     = (__bf16*)alloc((size_t)TD * TD * 2);
    __bf16* Wff1T   = (__bf16*)alloc((size_t)TF * TD * 2);
    __bf16* Wff2T   = (__bf16*)alloc((size_t)TD * TF * 2);
    __bf16* WoutT   = (__bf16*)alloc((size_t)TVPAD * TD * 2);
    float*  qkv_f32 = (float*) alloc((size_t)TM * 3 * TD * 4);
    __bf16* qbf     = (__bf16*)alloc((size_t)TM * TD * 2);
    __bf16* kbf     = (__bf16*)alloc((size_t)TM * TD * 2);
    __bf16* vTbf    = (__bf16*)alloc((size_t)TM * TD * 2);
    __bf16* attn_bf = (__bf16*)alloc((size_t)TM * TD * 2);
    float*  mha_f32 = (float*) alloc((size_t)TM * TD * 4);
    float*  h1_f32  = (float*) alloc((size_t)TM * TD * 4);
    __bf16* h1_bf   = (__bf16*)alloc((size_t)TM * TD * 2);
    __bf16* ff1_bf  = (__bf16*)alloc((size_t)TM * TF * 2);
    float*  s2_f32  = (float*) alloc((size_t)TM * TD * 4);
    __bf16* h2_bf   = (__bf16*)alloc((size_t)TM * TD * 2);

    // 1) weight prep (bf16, transposed [N,K])
    wtrans_kernel<<<(3 * TD * TD) / 256, 256, 0, stream>>>(W_qkv, WqkvT, TD, 3 * TD, 3 * TD);
    wosum_kernel<<<(TD * TD) / 256, 256, 0, stream>>>(W_o, WoT);
    wtrans_kernel<<<(TF * TD) / 256, 256, 0, stream>>>(W_ff1, Wff1T, TD, TF, TF);
    wtrans_kernel<<<(TD * TF) / 256, 256, 0, stream>>>(W_ff2, Wff2T, TF, TD, TD);
    wtrans_kernel<<<((size_t)TVPAD * TD) / 256, 256, 0, stream>>>(W_out, WoutT, TD, TV, TVPAD);

    // 2) embedding + positional encoding
    embed_kernel<<<((size_t)TM * TD) / 256, 256, 0, stream>>>(inputs, emb, x_f32, x_bf);

    // 3) QKV projection: [4096,768] @ [768,2304]
    gemm_bf16_kernel<<<dim3((3 * TD) / 256, TM / 64), 256, 0, stream>>>(
        x_bf, WqkvT, b_qkv, nullptr, qkv_f32, nullptr, TM, 3 * TD, TD, 0);

    // 4) split to q (scaled), k, vT
    split_qkv_kernel<<<((size_t)TM * TD) / 256, 256, 0, stream>>>(qkv_f32, qbf, kbf, vTbf);

    // 5) causal flash attention
    attention_kernel<<<dim3(TS / 128, TH, TB), 256, 0, stream>>>(qbf, kbf, vTbf, attn_bf);

    // 6) output projection (W_o pre-summed over tiled heads) + residual x
    gemm_bf16_kernel<<<dim3(TD / 256, TM / 64), 256, 0, stream>>>(
        attn_bf, WoT, b_o, x_f32, mha_f32, nullptr, TM, TD, TD, 0);

    // 7) LN1
    layernorm_kernel<<<TM, 256, 0, stream>>>(mha_f32, ln1_g, ln1_b, h1_f32, h1_bf);

    // 8) FF1 + ReLU (bf16 out)
    gemm_bf16_kernel<<<dim3(TF / 256, TM / 64), 256, 0, stream>>>(
        h1_bf, Wff1T, b_ff1, nullptr, nullptr, ff1_bf, TM, TF, TD, 1);

    // 9) FF2 + residual h1
    gemm_bf16_kernel<<<dim3(TD / 256, TM / 64), 256, 0, stream>>>(
        ff1_bf, Wff2T, b_ff2, h1_f32, s2_f32, nullptr, TM, TD, TF, 0);

    // 10) LN2
    layernorm_kernel<<<TM, 256, 0, stream>>>(s2_f32, ln2_g, ln2_b, nullptr, h2_bf);

    // 11) vocab projection -> d_out (f32, true N = 50257)
    gemm_bf16_kernel<<<dim3(TVPAD / 256, TM / 64), 256, 0, stream>>>(
        h2_bf, WoutT, b_out, nullptr, out, nullptr, TM, TV, TD, 0);
}
